// CIM_34677565948716
// MI455X (gfx1250) — matrix operand order repaired and checked by reading, not compiled, not run on hardware
//
#include <hip/hip_runtime.h>

#if __has_include(<hip/amd_detail/amd_gfx1250_TDM.h>)
#define ATHENA_TDM_6ARG 1
#endif

typedef __attribute__((ext_vector_type(16))) __bf16 v16bf;
typedef __attribute__((ext_vector_type(8)))  __bf16 v8bf;
typedef __attribute__((ext_vector_type(8)))  float  v8f;
typedef __attribute__((ext_vector_type(4))) unsigned int v4u;
typedef __attribute__((ext_vector_type(8)))  int v8i;
typedef __attribute__((ext_vector_type(4)))  int v4i;

union BF16Frag { v16bf v; v8bf h[2]; };

#define BN_EPS 1e-5f

static __device__ __forceinline__ v8f wmma_bf16(v16bf a, v16bf b, v8f c) {
  return __builtin_amdgcn_wmma_f32_16x16x32_bf16(false, a, false, b, (short)0, c, false, false);
}

// A operand (16x32 bf16): lane holds row; element order K = {kb..kb+7, kb+16..kb+23}, kb=(lane>=16)*8
static __device__ __forceinline__ v16bf load_a_frag(const __bf16* rowp, int lane) {
  int kb = (lane >> 4) * 8;
  BF16Frag f;
  f.h[0] = *(const v8bf*)(rowp + kb);
  f.h[1] = *(const v8bf*)(rowp + kb + 16);
  return f.v;
}

// B operand (32x16 bf16): lane holds column; element order K = {kb2..kb2+7, kb2+16..kb2+23}, kb2=(lane>=16)*8
static __device__ __forceinline__ v16bf load_b_frag(const __bf16* colp, int lane) {
  int kb2 = (lane >> 4) * 8;
  BF16Frag f;
  f.h[0] = *(const v8bf*)(colp + kb2);
  f.h[1] = *(const v8bf*)(colp + kb2 + 16);
  return f.v;
}

static __device__ __forceinline__ float xor16(float x) {
  int i = __builtin_bit_cast(int, x);
  i = __builtin_amdgcn_ds_swizzle(i, 0x401F);  // group-of-32, xor_mask=0x10
  return __builtin_bit_cast(float, i);
}

static __device__ __forceinline__ void lds_fence_wave() {
  __builtin_amdgcn_wave_barrier();
  asm volatile("s_wait_dscnt 0x0" ::: "memory");
  __builtin_amdgcn_wave_barrier();
}

// ---------------------------------------------------------------------------
// TDM: DMA a 2D bf16 tile [rows x 32 elems], row stride 256 elems, into LDS.
// D# per CDNA5 ISA ch.8: group0 = {count, lds_addr, global_addr, type=2},
// group1 = {data_size=2B, tensor dims, tile dims, dim0 stride}.
// ---------------------------------------------------------------------------
static __device__ __forceinline__ void tdm_load_tile_32xR(
    const __bf16* gptr, unsigned lds_off, unsigned rows) {
  unsigned long long ga = (unsigned long long)(uintptr_t)gptr;
  v4u g0;
  g0[0] = 1u;                                   // count=1, user descriptor
  g0[1] = lds_off;                              // LDS byte address
  g0[2] = (unsigned)(ga & 0xffffffffu);         // global_addr[31:0]
  g0[3] = (unsigned)((ga >> 32) & 0x01ffffffu)  // global_addr[56:32]
          | (2u << 30);                         // type = 2 ("image")
  v8i g1;
  g1[0] = (int)(1u << 16);                      // data_size = 1 (2 bytes)
  g1[1] = (int)(32u << 16);                     // tensor_dim0 = 32 (low 16 into [47:32]=0)
  g1[2] = (int)(rows << 16);                    // tensor_dim0 hi=0 | tensor_dim1 lo
  g1[3] = (int)(32u << 16);                     // tensor_dim1 hi=0 | tile_dim0 = 32
  g1[4] = (int)rows;                            // tile_dim1 = rows, tile_dim2 = 0
  g1[5] = 256;                                  // tensor_dim0_stride = 256 elems
  g1[6] = 0;
  g1[7] = 0;
  v4i z4 = {0, 0, 0, 0};
#ifdef ATHENA_TDM_6ARG
  v8i z8 = {0, 0, 0, 0, 0, 0, 0, 0};
  __builtin_amdgcn_tensor_load_to_lds(g0, g1, z4, z4, z8, 0);
#else
  __builtin_amdgcn_tensor_load_to_lds(g0, g1, z4, z4, 0);
#endif
}

// ---------------------------------------------------------------------------
// Transpose-convert: f32 [B][Cc][Nn] (channel-first) -> bf16 [B][Nn][Cc] (channel-last)
// grid: (Nn/32, Cc/32, B), block 256
// ---------------------------------------------------------------------------
__global__ __launch_bounds__(256) void cvt_transpose_kernel(
    const float* __restrict__ in, __bf16* __restrict__ out, int Cc, int Nn) {
  __shared__ float tile[32][33];
  int b = blockIdx.z;
  int n0 = blockIdx.x * 32, c0 = blockIdx.y * 32;
  int tx = threadIdx.x & 31, ty = threadIdx.x >> 5;
#pragma unroll
  for (int i = 0; i < 32; i += 8)
    tile[ty + i][tx] = in[((size_t)b * Cc + c0 + ty + i) * Nn + n0 + tx];
  __syncthreads();
#pragma unroll
  for (int i = 0; i < 32; i += 8)
    out[((size_t)b * Nn + n0 + ty + i) * Cc + c0 + tx] = (__bf16)tile[tx][ty + i];
}

// ---------------------------------------------------------------------------
// Fold BN0 into w_q (elementwise) + folded bias; also convert w_out to bf16.
// grid: 256 blocks (one per o), block 256 (one per c)
// ---------------------------------------------------------------------------
__global__ __launch_bounds__(256) void fold_q_out_kernel(
    const float* __restrict__ w_q, const float* __restrict__ b_q,
    const float* __restrict__ g0, const float* __restrict__ be0,
    const float* __restrict__ mn0, const float* __restrict__ vr0,
    const float* __restrict__ w_out,
    __bf16* __restrict__ wq_eff, float* __restrict__ bq_eff,
    __bf16* __restrict__ wout_bf) {
  __shared__ float red[256];
  int o = blockIdx.x, c = threadIdx.x;
  float a0 = g0[c] * rsqrtf(vr0[c] + BN_EPS);
  float c0 = be0[c] - mn0[c] * a0;
  float wq = w_q[o * 256 + c];
  wq_eff[o * 256 + c] = (__bf16)(wq * a0);
  wout_bf[o * 256 + c] = (__bf16)w_out[o * 256 + c];
  red[c] = wq * c0;
  __syncthreads();
  for (int s = 128; s > 0; s >>= 1) { if (c < s) red[c] += red[c + s]; __syncthreads(); }
  if (c == 0) bq_eff[o] = b_q[o] + red[0];
}

// ---------------------------------------------------------------------------
// Compose: w_eff[o][j] = sum_c w_x[o][c]*a[c]*w_ht[c][j]  (fold BN + h->h_al conv)
// grid: (512/128, 256), block 128
// ---------------------------------------------------------------------------
__global__ __launch_bounds__(128) void fold_kv_w_kernel(
    const float* __restrict__ w_x, const float* __restrict__ w_ht,
    const float* __restrict__ g, const float* __restrict__ vr,
    __bf16* __restrict__ w_eff) {
  int o = blockIdx.y;
  int j = blockIdx.x * 128 + threadIdx.x;
  float acc = 0.f;
  for (int c = 0; c < 256; ++c) {
    float a = g[c] * rsqrtf(vr[c] + BN_EPS);
    acc += w_x[o * 256 + c] * a * w_ht[c * 512 + j];
  }
  w_eff[o * 512 + j] = (__bf16)acc;
}

__global__ __launch_bounds__(64) void fold_kv_b_kernel(
    const float* __restrict__ w_x, const float* __restrict__ b_x,
    const float* __restrict__ b_ht,
    const float* __restrict__ g, const float* __restrict__ be,
    const float* __restrict__ mn, const float* __restrict__ vr,
    float* __restrict__ b_eff) {
  int o = blockIdx.x * 64 + threadIdx.x;
  float acc = b_x[o];
  for (int c = 0; c < 256; ++c) {
    float a = g[c] * rsqrtf(vr[c] + BN_EPS);
    float cc = be[c] - mn[c] * a;
    float w = w_x[o * 256 + c];
    acc += w * cc + w * a * b_ht[c];
  }
  b_eff[o] = acc;
}

// ---------------------------------------------------------------------------
// GEMM (channel-last out): Y[b][n][o] = bias[o] + sum_c X[b][n][c] * W[o][c]
// A = X rows, B = W rows (as columns of W^T). Y bf16.
// grid: (Nrows/64, O/128, B), block 128 (4 waves, 2n x 2o)
// ---------------------------------------------------------------------------
__global__ __launch_bounds__(128) void gemm_ncl_kernel(
    const __bf16* __restrict__ X, const __bf16* __restrict__ W,
    const float* __restrict__ bias, __bf16* __restrict__ Y,
    int Nrows, int K) {
  int b = blockIdx.z;
  int tid = threadIdx.x, wave = tid >> 5, lane = tid & 31;
  int n_base = blockIdx.x * 64 + (wave >> 1) * 32;
  int o_base = blockIdx.y * 128 + (wave & 1) * 64;
  int lanelo = lane & 15;
  v8f acc[2][4] = {};
  for (int kk = 0; kk < K; kk += 32) {
    v16bf af[2], bfr[4];
#pragma unroll
    for (int nt = 0; nt < 2; ++nt) {
      const __bf16* xp = X + ((size_t)b * Nrows + n_base + nt * 16 + lanelo) * K + kk;
      af[nt] = load_a_frag(xp, lane);
      if (kk + 32 < K) __builtin_prefetch(xp + 32, 0, 1);  // global_prefetch_b8
    }
#pragma unroll
    for (int ot = 0; ot < 4; ++ot)
      bfr[ot] = load_b_frag(W + (size_t)(o_base + ot * 16 + lanelo) * K + kk, lane);
#pragma unroll
    for (int nt = 0; nt < 2; ++nt)
#pragma unroll
      for (int ot = 0; ot < 4; ++ot)
        acc[nt][ot] = wmma_bf16(af[nt], bfr[ot], acc[nt][ot]);
  }
#pragma unroll
  for (int ot = 0; ot < 4; ++ot) {
    int o = o_base + ot * 16 + lanelo;
    float bv = bias[o];
#pragma unroll
    for (int nt = 0; nt < 2; ++nt) {
#pragma unroll
      for (int r = 0; r < 8; ++r) {
        int n = n_base + nt * 16 + r + (lane >> 4) * 8;
        Y[((size_t)b * Nrows + n) * 256 + o] = (__bf16)(acc[nt][ot][r] + bv);
      }
    }
  }
}

// ---------------------------------------------------------------------------
// Fused multi-head attention, flash-style, per (b,head).
// q: bf16 [B][4096][256], k/v: bf16 [B][1024][256], out: bf16 [B][4096][256]
// grid: (B*HEADS, 8), block 128 (4 waves). Each wave: 8 strips of 16 q-rows.
// k-head is DMA'd into LDS by the Tensor Data Mover; v-head is transposed
// into LDS manually (TDM cannot transpose).
// ---------------------------------------------------------------------------
__global__ __launch_bounds__(128) void attn_kernel(
    const __bf16* __restrict__ q, const __bf16* __restrict__ k,
    const __bf16* __restrict__ v, __bf16* __restrict__ attn_out) {
  __shared__ __bf16 kBuf[1024 * 32];      // 64 KB: k head [m][d], filled by TDM
  __shared__ __bf16 vT[32 * 1024];        // 64 KB: v head transposed [d][m]
  __shared__ float  sS[4][16 * 64];       // 16 KB: per-wave score chunk
  __shared__ __bf16 pP[4][16 * 64];       //  8 KB: per-wave P chunk (A layout)
  __shared__ float  facS[4][16];
  __shared__ float  lS[4][16];

  int bh = blockIdx.x, b = bh >> 3, hh = bh & 7;
  int tid = threadIdx.x, wave = tid >> 5, lane = tid & 31;
  int lanelo = lane & 15, half = lane >> 4;
  const float scale = 0.17677669529663687f;  // 1/sqrt(32)

  // kick off TDM: k[b, :, hh*32 .. hh*32+31] -> kBuf  (wave 0 owns TENSORcnt)
  if (tid < 32) {
    tdm_load_tile_32xR(k + (size_t)b * 1024 * 256 + hh * 32,
                       (unsigned)(uintptr_t)&kBuf[0], 1024u);
  }
  // stage v_head transposed into LDS (overlaps with the TDM transfer)
  const __bf16* vg = v + (size_t)b * 1024 * 256 + hh * 32;
  for (int idx = tid; idx < 32 * 1024; idx += 128) {
    int m = idx >> 5, d = idx & 31;
    vT[d * 1024 + m] = vg[(size_t)m * 256 + d];
  }
  if (tid < 32) __builtin_amdgcn_s_wait_tensorcnt(0);
  __syncthreads();

  float* sSw = &sS[wave][0];
  __bf16* pPw = &pP[wave][0];

  for (int si = 0; si < 8; ++si) {
    int strip = blockIdx.y * 32 + wave * 8 + si;
    int n0 = strip * 16;
    v16bf aq = load_a_frag(q + ((size_t)b * 4096 + n0 + lanelo) * 256 + hh * 32, lane);
    v8f O0 = {}, O1 = {};
    float mrun = -1e30f, lrun = 0.f;
    int srow = lanelo;

    for (int mc = 0; mc < 1024; mc += 64) {
      // ---- S = q k^T for this 16x64 chunk (4 WMMAs, k from LDS) ----
#pragma unroll
      for (int mt = 0; mt < 4; ++mt) {
        int mcol = mc + mt * 16 + lanelo;
        v16bf bk = load_b_frag(kBuf + (size_t)mcol * 32, lane);
        v8f s = {};
        s = wmma_bf16(aq, bk, s);
#pragma unroll
        for (int r = 0; r < 8; ++r)
          sSw[(r + half * 8) * 64 + mt * 16 + lanelo] = s[r];
      }
      lds_fence_wave();

      // ---- online softmax: lane owns (row=lane&15, column half) ----
      float sv[32];
      float cmax = -1e30f;
#pragma unroll
      for (int j = 0; j < 32; ++j) {
        sv[j] = sSw[srow * 64 + half * 32 + j] * scale;
        cmax = fmaxf(cmax, sv[j]);
      }
      cmax = fmaxf(cmax, xor16(cmax));
      float mnew = fmaxf(mrun, cmax);
      float psum = 0.f;
#pragma unroll
      for (int j = 0; j < 32; ++j) {
        float p = __expf(sv[j] - mnew);
        psum += p;
        pPw[srow * 64 + half * 32 + j] = (__bf16)p;
      }
      psum += xor16(psum);
      float fr = __expf(mrun - mnew);
      lrun = lrun * fr + psum;
      mrun = mnew;
      if (half == 0) facS[wave][srow] = fr;
      lds_fence_wave();

      // ---- rescale running O ----
#pragma unroll
      for (int r = 0; r < 8; ++r) {
        float ff = facS[wave][r + half * 8];
        O0[r] *= ff; O1[r] *= ff;
      }
      // ---- O += P * V (4 WMMAs) ----
#pragma unroll
      for (int kt = 0; kt < 2; ++kt) {
        v16bf ap = load_a_frag(pPw + lanelo * 64 + kt * 32, lane);
        v16bf bv0 = load_b_frag(vT + (size_t)(0 + lanelo) * 1024 + mc + kt * 32, lane);
        v16bf bv1 = load_b_frag(vT + (size_t)(16 + lanelo) * 1024 + mc + kt * 32, lane);
        O0 = wmma_bf16(ap, bv0, O0);
        O1 = wmma_bf16(ap, bv1, O1);
      }
      lds_fence_wave();
    }

    // ---- epilogue: divide by l, store channel-last bf16 ----
    if (half == 0) lS[wave][srow] = lrun;
    lds_fence_wave();
    __bf16* og = attn_out + (size_t)b * 4096 * 256 + hh * 32;
#pragma unroll
    for (int r = 0; r < 8; ++r) {
      float inv = 1.0f / lS[wave][r + half * 8];
      int n = n0 + r + half * 8;
      og[(size_t)n * 256 + lanelo]      = (__bf16)(O0[r] * inv);
      og[(size_t)n * 256 + 16 + lanelo] = (__bf16)(O1[r] * inv);
    }
  }
}

// ---------------------------------------------------------------------------
// Final projection (channel-first out + bias + residual):
// out[b][o][n] = sum_c W[o][c]*X[b][n][c] + bias[o] + R[b][o][n]
// grid: (256/128, 4096/64, B), block 128 (4 waves, 2o x 2n)
// ---------------------------------------------------------------------------
__global__ __launch_bounds__(128) void final_proj_kernel(
    const __bf16* __restrict__ W, const __bf16* __restrict__ X,
    const float* __restrict__ bias, const float* __restrict__ R,
    float* __restrict__ out) {
  int b = blockIdx.z;
  int tid = threadIdx.x, wave = tid >> 5, lane = tid & 31;
  int o_base = blockIdx.x * 128 + (wave >> 1) * 64;
  int n_base = blockIdx.y * 64 + (wave & 1) * 32;
  int lanelo = lane & 15;
  v8f acc[4][2] = {};
  for (int kk = 0; kk < 256; kk += 32) {
    v16bf af[4], bfr[2];
#pragma unroll
    for (int ot = 0; ot < 4; ++ot)
      af[ot] = load_a_frag(W + (size_t)(o_base + ot * 16 + lanelo) * 256 + kk, lane);
#pragma unroll
    for (int nt = 0; nt < 2; ++nt) {
      const __bf16* xp = X + ((size_t)b * 4096 + n_base + nt * 16 + lanelo) * 256 + kk;
      bfr[nt] = load_b_frag(xp, lane);
      if (kk + 32 < 256) __builtin_prefetch(xp + 32, 0, 1);
    }
#pragma unroll
    for (int ot = 0; ot < 4; ++ot)
#pragma unroll
      for (int nt = 0; nt < 2; ++nt)
        acc[ot][nt] = wmma_bf16(af[ot], bfr[nt], acc[ot][nt]);
  }
#pragma unroll
  for (int ot = 0; ot < 4; ++ot) {
#pragma unroll
    for (int r = 0; r < 8; ++r) {
      int o = o_base + ot * 16 + r + (lane >> 4) * 8;
      float bv = bias[o];
#pragma unroll
      for (int nt = 0; nt < 2; ++nt) {
        int n = n_base + nt * 16 + lanelo;
        size_t idx = ((size_t)b * 256 + o) * 4096 + n;
        out[idx] = acc[ot][nt][r] + bv + R[idx];
      }
    }
  }
}

// ---------------------------------------------------------------------------
extern "C" void kernel_launch(void* const* d_in, const int* in_sizes, int n_in,
                              void* d_out, int out_size, void* d_ws, size_t ws_size,
                              hipStream_t stream) {
  (void)in_sizes; (void)n_in; (void)out_size; (void)ws_size;
  const float* f     = (const float*)d_in[0];
  const float* h     = (const float*)d_in[1];
  const float* w_ht  = (const float*)d_in[2];
  const float* b_ht  = (const float*)d_in[3];
  const float* bn_g  = (const float*)d_in[4];   // (3,256)
  const float* bn_b  = (const float*)d_in[5];
  const float* bn_m  = (const float*)d_in[6];
  const float* bn_v  = (const float*)d_in[7];
  const float* w_q   = (const float*)d_in[8];
  const float* b_q   = (const float*)d_in[9];
  const float* w_k   = (const float*)d_in[10];
  const float* b_k   = (const float*)d_in[11];
  const float* w_v   = (const float*)d_in[12];
  const float* b_v   = (const float*)d_in[13];
  const float* w_out = (const float*)d_in[14];
  const float* b_out = (const float*)d_in[15];
  float* out = (float*)d_out;

  char* ws = (char*)d_ws;
  __bf16* fT   = (__bf16*)(ws);                       // 4*4096*256 bf16 = 8 MiB
  __bf16* hT   = (__bf16*)(ws + (8u << 20));          // 4*1024*512 bf16 = 4 MiB
  __bf16* qb   = (__bf16*)(ws + (12u << 20));         // 8 MiB
  __bf16* kb   = (__bf16*)(ws + (20u << 20));         // 2 MiB
  __bf16* vb   = (__bf16*)(ws + (22u << 20));         // 2 MiB
  __bf16* ab   = (__bf16*)(ws + (24u << 20));         // 8 MiB
  __bf16* wqE  = (__bf16*)(ws + (32u << 20));         // 128 KiB
  __bf16* wkE  = (__bf16*)(ws + (32u << 20) + 131072);        // 256 KiB
  __bf16* wvE  = (__bf16*)(ws + (32u << 20) + 131072 + 262144);
  __bf16* woB  = (__bf16*)(ws + (32u << 20) + 131072 + 2 * 262144);
  float*  bqE  = (float*)(ws + (32u << 20) + 2 * 131072 + 2 * 262144);
  float*  bkE  = bqE + 256;
  float*  bvE  = bkE + 256;

  // 1) channel-last bf16 copies of f and h
  cvt_transpose_kernel<<<dim3(128, 8, 4), 256, 0, stream>>>(f, fT, 256, 4096);
  cvt_transpose_kernel<<<dim3(32, 16, 4), 256, 0, stream>>>(h, hT, 512, 1024);

  // 2) fold BN (and h->h_al conv) into effective weights/biases
  fold_q_out_kernel<<<256, 256, 0, stream>>>(w_q, b_q,
      bn_g + 0, bn_b + 0, bn_m + 0, bn_v + 0, w_out, wqE, bqE, woB);
  fold_kv_w_kernel<<<dim3(4, 256), 128, 0, stream>>>(w_k, w_ht, bn_g + 256, bn_v + 256, wkE);
  fold_kv_w_kernel<<<dim3(4, 256), 128, 0, stream>>>(w_v, w_ht, bn_g + 512, bn_v + 512, wvE);
  fold_kv_b_kernel<<<4, 64, 0, stream>>>(w_k, b_k, b_ht, bn_g + 256, bn_b + 256, bn_m + 256, bn_v + 256, bkE);
  fold_kv_b_kernel<<<4, 64, 0, stream>>>(w_v, b_v, b_ht, bn_g + 512, bn_b + 512, bn_m + 512, bn_v + 512, bvE);

  // 3) Q/K/V projections (WMMA), channel-last bf16 outputs
  gemm_ncl_kernel<<<dim3(64, 2, 4), 128, 0, stream>>>(fT, wqE, bqE, qb, 4096, 256);
  gemm_ncl_kernel<<<dim3(16, 2, 4), 128, 0, stream>>>(hT, wkE, bkE, kb, 1024, 512);
  gemm_ncl_kernel<<<dim3(16, 2, 4), 128, 0, stream>>>(hT, wvE, bvE, vb, 1024, 512);

  // 4) fused flash attention per (b, head); k staged via Tensor Data Mover
  attn_kernel<<<dim3(32, 8), 128, 0, stream>>>(qb, kb, vb, ab);

  // 5) output projection + bias + residual, channel-first f32
  final_proj_kernel<<<dim3(2, 64, 4), 128, 0, stream>>>(woB, ab, b_out, f, out);
}
